// PerceiverAttention_72670846648790
// MI455X (gfx1250) — matrix operand order repaired and checked by reading, not compiled, not run on hardware
//
#include <hip/hip_runtime.h>
#include <hip/hip_bf16.h>

// ---------------------------------------------------------------------------
// PerceiverAttention for MI455X (gfx1250, wave32, WMMA bf16 16x16x32)
// Pipeline: LN+cast -> GEMM(kv) -> GEMM(q) -> flash-attn -> GEMM(out, f32)
// Round 4: TDM builtin uses the 6-arg (clang-23) form:
//   (u32x4 g0, i32x8 g1, i32x4 g2, i32x4 g3, i32x8 g4, i32 cpol)
// GEMM A tiles keep the async-DMA path (global_load_async_to_lds_b128).
// ---------------------------------------------------------------------------

typedef __attribute__((ext_vector_type(16))) __bf16 bf16x16;
typedef __attribute__((ext_vector_type(8)))  float  f32x8;
typedef __attribute__((ext_vector_type(4)))  unsigned int u32x4;
typedef __attribute__((ext_vector_type(8)))  int  i32x8;
typedef __attribute__((ext_vector_type(4)))  int  i32x4;

#define WMMA_BF16(A, B, C) \
  __builtin_amdgcn_wmma_f32_16x16x32_bf16(false, (A), false, (B), (short)0, (C), false, false)

#if __has_builtin(__builtin_amdgcn_tensor_load_to_lds)
#define HAVE_TDM 1
#else
#define HAVE_TDM 0
#endif

union Frag  { bf16x16 v; uint4 q[2]; };
union OutPk { __bf16 h[4]; uint2 u; };

// Flat shared-aperture pointers carry the wave-relative LDS byte offset in
// their low 32 bits (ISA 10.2).
__device__ __forceinline__ uint32_t lds_offset(const void* p) {
  return (uint32_t)(uintptr_t)p;
}

// ---------------------------------------------------------------------------
// LayerNorm over D=1024 + cast to bf16. One block (256 thr) per row.
// ---------------------------------------------------------------------------
__global__ __launch_bounds__(256)
void ln_cast_kernel(const float* __restrict__ x, const float* __restrict__ g,
                    const float* __restrict__ beta, __bf16* __restrict__ out) {
  constexpr int D = 1024;
  __shared__ float red[8];
  const size_t row = blockIdx.x;
  const int t = threadIdx.x;
  const float* xr = x + row * (size_t)D;
  float4 v = *(const float4*)(xr + t * 4);
  float s = v.x + v.y + v.z + v.w;
#pragma unroll
  for (int m = 16; m >= 1; m >>= 1) s += __shfl_xor(s, m, 32);
  if ((t & 31) == 0) red[t >> 5] = s;
  __syncthreads();
  float tot = 0.f;
#pragma unroll
  for (int i = 0; i < 8; ++i) tot += red[i];
  const float mu = tot * (1.0f / D);
  __syncthreads();
  const float dx = v.x - mu, dy = v.y - mu, dz = v.z - mu, dw = v.w - mu;
  float s2 = dx * dx + dy * dy + dz * dz + dw * dw;
#pragma unroll
  for (int m = 16; m >= 1; m >>= 1) s2 += __shfl_xor(s2, m, 32);
  if ((t & 31) == 0) red[t >> 5] = s2;
  __syncthreads();
  float tot2 = 0.f;
#pragma unroll
  for (int i = 0; i < 8; ++i) tot2 += red[i];
  const float rs = rsqrtf(tot2 * (1.0f / D) + 1e-5f);
  float4 gv = *(const float4*)(g + t * 4);
  float4 bv = *(const float4*)(beta + t * 4);
  OutPk o;
  o.h[0] = (__bf16)(dx * rs * gv.x + bv.x);
  o.h[1] = (__bf16)(dy * rs * gv.y + bv.y);
  o.h[2] = (__bf16)(dz * rs * gv.z + bv.z);
  o.h[3] = (__bf16)(dw * rs * gv.w + bv.w);
  *(uint2*)(out + row * (size_t)D + t * 4) = o.u;
}

// ---------------------------------------------------------------------------
// fp32 -> bf16 elementwise (weights), vectorized x4
// ---------------------------------------------------------------------------
__global__ __launch_bounds__(256)
void cast_kernel(const float* __restrict__ in, __bf16* __restrict__ out, int n4) {
  const int i = blockIdx.x * blockDim.x + threadIdx.x;
  if (i >= n4) return;
  float4 v = ((const float4*)in)[i];
  OutPk o;
  o.h[0] = (__bf16)v.x; o.h[1] = (__bf16)v.y;
  o.h[2] = (__bf16)v.z; o.h[3] = (__bf16)v.w;
  ((uint2*)out)[i] = o.u;
}

// ---------------------------------------------------------------------------
// C[M,N] = A[M,K] @ B[K,N], bf16 in, f32 accumulate via WMMA.
// Workgroup tile 64x64 (4 waves; wave w -> rows [w*16,w*16+16) x 64 cols).
// A tile: ASYNC DMA straight to LDS (ASYNCcnt-tracked, no VGPR bounce).
// B tile: VGPR path (needs transpose into Bt[n][k] during staging).
// ---------------------------------------------------------------------------
template <bool OUT_F32>
__global__ __launch_bounds__(128)
void gemm_wmma_kernel(const __bf16* __restrict__ A, const __bf16* __restrict__ B,
                      void* __restrict__ C, int M, int N, int K) {
  __shared__ __bf16 As[64][32];
  __shared__ __bf16 Bt[64][32];
  const int tid = threadIdx.x;
  const int lane = tid & 31;
  const int wave = tid >> 5;
  const int r = lane & 15;
  const int half = lane >> 4;
  const int m0 = blockIdx.y * 64;
  const int n0 = blockIdx.x * 64;
  (void)M;
  f32x8 acc[4] = {};
  for (int kk = 0; kk < K; kk += 32) {
    const int idx = tid * 8;
#pragma unroll
    for (int i = 0; i < 2; ++i) {
      const int e = idx + i * 1024;
      // A tile 64x32: async DMA 16B per lane directly into LDS
      const int arow = e >> 5, acol = e & 31;
      const uint32_t lds_a = lds_offset(&As[arow][acol]);
      const unsigned long long ga =
          (unsigned long long)(uintptr_t)&A[(size_t)(m0 + arow) * K + kk + acol];
      asm volatile("global_load_async_to_lds_b128 %0, %1, off"
                   :: "v"(lds_a), "v"(ga) : "memory");
      // B tile 32x64: load 16B, scatter transposed into Bt[n][k]
      const int brow = e >> 6, bcol = e & 63;
      uint4 d = *(const uint4*)&B[(size_t)(kk + brow) * N + n0 + bcol];
      const __bf16* dv = (const __bf16*)&d;
#pragma unroll
      for (int j = 0; j < 8; ++j) Bt[bcol + j][brow] = dv[j];
    }
    if (kk + 32 < K)  // pull next B slab toward L2 (global_prefetch_b8)
      __builtin_prefetch(&B[(size_t)(kk + 32 + (idx >> 6)) * N + n0 + (idx & 63)], 0, 0);
    asm volatile("s_wait_asynccnt 0x0" ::: "memory");  // drain async DMA -> LDS
    __syncthreads();
    Frag a;
    a.q[0] = *(const uint4*)&As[wave * 16 + r][half * 8];
    a.q[1] = *(const uint4*)&As[wave * 16 + r][half * 8 + 16];
#pragma unroll
    for (int j = 0; j < 4; ++j) {
      Frag b;
      b.q[0] = *(const uint4*)&Bt[j * 16 + r][half * 8];
      b.q[1] = *(const uint4*)&Bt[j * 16 + r][half * 8 + 16];
      acc[j] = WMMA_BF16(a.v, b.v, acc[j]);
    }
    __syncthreads();
  }
  // C layout: lanes0-15 vgpr i -> M=i; lanes16-31 vgpr i -> M=8+i; N = lane&15
  const int crow = m0 + wave * 16 + half * 8;
  const int ccol = n0 + r;
#pragma unroll
  for (int j = 0; j < 4; ++j) {
#pragma unroll
    for (int i = 0; i < 8; ++i) {
      const size_t off = (size_t)(crow + i) * N + ccol + j * 16;
      if (OUT_F32) ((float*)C)[off] = acc[j][i];
      else         ((__bf16*)C)[off] = (__bf16)acc[j][i];
    }
  }
}

// ---------------------------------------------------------------------------
// Flash attention: grid = 64 (b*h), block = 128 (4 waves), wave -> 16 q rows.
// q: [512][512] bf16, kv: [65536][1024] bf16 (k cols 0..511, v 512..1023).
// Per 32-key chunk:
//   - K slab (32 keys x 64 dims) fetched by the TDM into LDS (one 2D D#
//     descriptor per chunk, issued by wave 0, TENSORcnt-drained),
//   - V slab staged transposed via VGPRs,
//   - S via WMMA, online softmax with 16-lane xor-reductions,
//   - P routed through LDS (C-layout -> A-layout), O += P @ V via WMMA.
// ---------------------------------------------------------------------------
__global__ __launch_bounds__(128)
void attn_kernel(const __bf16* __restrict__ q, const __bf16* __restrict__ kv,
                 __bf16* __restrict__ o) {
  const int bh = blockIdx.x;
  const int batch = bh >> 3;
  const int head = bh & 7;
  const int tid = threadIdx.x;
  const int lane = tid & 31;
  const int wave = tid >> 5;
  const int r = lane & 15;
  const int half = lane >> 4;
  __shared__ __bf16 Krm[32][64];     // [key][d] row-major K slab (TDM dest)
  __shared__ __bf16 Vt[64][32];      // [d][key] transposed V slab
  __shared__ __bf16 Ps[4][16][32];   // per-wave P (16 q rows x 32 keys)

  Frag qa[2];
  const __bf16* qp = q + (size_t)(batch * 64 + wave * 16 + r) * 512 + head * 64;
#pragma unroll
  for (int s = 0; s < 2; ++s) {
    qa[s].q[0] = *(const uint4*)(qp + s * 32 + half * 8);
    qa[s].q[1] = *(const uint4*)(qp + s * 32 + half * 8 + 16);
  }

  f32x8 oacc[4] = {};
  float m_run[8], l_run[8];
#pragma unroll
  for (int i = 0; i < 8; ++i) { m_run[i] = -3.0e38f; l_run[i] = 0.f; }
  const float scale = 0.125f;  // 64^-0.5

  const __bf16* kbase = kv + (size_t)batch * 8192 * 1024 + head * 64;
  const __bf16* vbase = kbase + 512;

  for (int j0 = 0; j0 < 8192; j0 += 32) {
#if HAVE_TDM
    // ---- TDM: one 2D descriptor moves the K slab (64 x 32 tile, 2B elems,
    //      row stride 1024 elems) straight into LDS. Wave 0 issues; the
    //      workgroup barrier below publishes it.
    if (wave == 0) {
      const unsigned long long ga =
          (unsigned long long)(uintptr_t)(kbase + (size_t)j0 * 1024);
      u32x4 g0;
      g0[0] = 1u;                                         // count=1 (valid D#)
      g0[1] = lds_offset(&Krm[0][0]);                     // lds_addr (bytes)
      g0[2] = (unsigned int)(ga & 0xFFFFFFFFu);           // global_addr[31:0]
      g0[3] = (unsigned int)((ga >> 32) & 0x1FFFFFFu)     // global_addr[56:32]
              | (2u << 30);                               // type=2 ("image")
      i32x8 g1;
      g1[0] = 0x10000;               // workgroup_mask=0, data_size=1 (2B)
      g1[1] = (int)(64u << 16);      // tensor_dim0 = 64 (low 16 in [31:16])
      g1[2] = (int)(8192u << 16);    // dim0 hi=0 | tensor_dim1 lo = 8192
      g1[3] = (int)(64u << 16);      // dim1 hi=0 | tile_dim0 = 64
      g1[4] = 32;                    // tile_dim1 = 32, tile_dim2 = 0
      g1[5] = 1024;                  // tensor_dim0_stride = 1024 elems (lo32)
      g1[6] = 0;                     // stride hi, dim1_stride lo
      g1[7] = 0;
      i32x4 g2 = {0, 0, 0, 0};       // unused (2D tensor)
      i32x4 g3 = {0, 0, 0, 0};
      i32x8 g4 = {0, 0, 0, 0, 0, 0, 0, 0};  // trailing group (6-arg form)
      __builtin_amdgcn_tensor_load_to_lds(g0, g1, g2, g3, g4, 0);
    }
#endif
    // stage V chunk (32 keys x 64 dims) transposed into LDS (overlaps TDM)
    const int idx = tid * 8;
#pragma unroll
    for (int i = 0; i < 2; ++i) {
      const int e = idx + i * 1024;
      const int krow = e >> 6;
      const int col = e & 63;
      uint4 d = *(const uint4*)(vbase + (size_t)(j0 + krow) * 1024 + col);
      const __bf16* dv = (const __bf16*)&d;
#pragma unroll
      for (int j = 0; j < 8; ++j) Vt[col + j][krow] = dv[j];
    }
#if HAVE_TDM
    if (wave == 0) __builtin_amdgcn_s_wait_tensorcnt((short)0);
#endif
    __syncthreads();

    // S = (q @ k^T) * scale, two 16-key tiles
    f32x8 sv[2];
#pragma unroll
    for (int t = 0; t < 2; ++t) {
      f32x8 c = {};
#pragma unroll
      for (int s = 0; s < 2; ++s) {
        Frag kb;  // B frag: col = key (lane&15), K = d in table order
#if HAVE_TDM
        const __bf16* kp = &Krm[t * 16 + r][0];
#else
        const __bf16* kp = kbase + (size_t)(j0 + t * 16 + r) * 1024;
#endif
        kb.q[0] = *(const uint4*)(kp + s * 32 + half * 8);
        kb.q[1] = *(const uint4*)(kp + s * 32 + half * 8 + 16);
        c = WMMA_BF16(qa[s].v, kb.v, c);
      }
      sv[t] = c * scale;
    }

    // online softmax over this chunk; each lane owns rows half*8 + i
#pragma unroll
    for (int i = 0; i < 8; ++i) {
      float mx = fmaxf(sv[0][i], sv[1][i]);
#pragma unroll
      for (int msk = 8; msk >= 1; msk >>= 1) mx = fmaxf(mx, __shfl_xor(mx, msk, 32));
      const float mnew = fmaxf(m_run[i], mx);
      const float alpha = __expf(m_run[i] - mnew);
      m_run[i] = mnew;
      l_run[i] *= alpha;
#pragma unroll
      for (int j = 0; j < 4; ++j) oacc[j][i] *= alpha;
#pragma unroll
      for (int t = 0; t < 2; ++t) {
        const float p = __expf(sv[t][i] - mnew);
        float ps = p;
#pragma unroll
        for (int msk = 8; msk >= 1; msk >>= 1) ps += __shfl_xor(ps, msk, 32);
        l_run[i] += ps;
        Ps[wave][half * 8 + i][t * 16 + r] = (__bf16)p;
      }
    }
    // P was written C-layout by all lanes of this wave; drain LDS before the
    // A-layout re-read (per-wave region, so a dscnt wait suffices, no barrier)
    asm volatile("s_wait_dscnt 0x0" ::: "memory");

    Frag pa;
    pa.q[0] = *(const uint4*)&Ps[wave][r][half * 8];
    pa.q[1] = *(const uint4*)&Ps[wave][r][half * 8 + 16];
#pragma unroll
    for (int j = 0; j < 4; ++j) {
      Frag vb;  // B frag: col = d (lane&15 within tile j), K = key in table order
      vb.q[0] = *(const uint4*)&Vt[j * 16 + r][half * 8];
      vb.q[1] = *(const uint4*)&Vt[j * 16 + r][half * 8 + 16];
      oacc[j] = WMMA_BF16(pa.v, vb.v, oacc[j]);
    }
    __syncthreads();
  }

  // normalize + write O
#pragma unroll
  for (int i = 0; i < 8; ++i) {
    const float inv = 1.0f / l_run[i];
    const size_t orow = (size_t)(batch * 64 + wave * 16 + half * 8 + i);
#pragma unroll
    for (int j = 0; j < 4; ++j) {
      o[orow * 512 + head * 64 + j * 16 + r] = (__bf16)(oacc[j][i] * inv);
    }
  }
}

// ---------------------------------------------------------------------------
extern "C" void kernel_launch(void* const* d_in, const int* in_sizes, int n_in,
                              void* d_out, int out_size, void* d_ws, size_t ws_size,
                              hipStream_t stream) {
  (void)in_sizes; (void)n_in; (void)out_size; (void)ws_size;
  const float* x    = (const float*)d_in[0];
  const float* lat  = (const float*)d_in[1];
  const float* g1   = (const float*)d_in[2];
  const float* b1   = (const float*)d_in[3];
  const float* g2   = (const float*)d_in[4];
  const float* b2   = (const float*)d_in[5];
  const float* Wq   = (const float*)d_in[6];
  const float* Wkv  = (const float*)d_in[7];
  const float* Wout = (const float*)d_in[8];
  float* out = (float*)d_out;

  char* ws = (char*)d_ws;
  __bf16* xn   = (__bf16*)ws; ws += (size_t)65536 * 1024 * 2;  // LN(x) bf16
  __bf16* kvb  = (__bf16*)ws; ws += (size_t)65536 * 1024 * 2;  // kv bf16
  __bf16* lnb  = (__bf16*)ws; ws += (size_t)512 * 1024 * 2;    // LN(latents)
  __bf16* wqb  = (__bf16*)ws; ws += (size_t)1024 * 512 * 2;
  __bf16* wkvb = (__bf16*)ws; ws += (size_t)1024 * 1024 * 2;
  __bf16* wob  = (__bf16*)ws; ws += (size_t)512 * 1024 * 2;
  __bf16* qb   = (__bf16*)ws; ws += (size_t)512 * 512 * 2;     // q
  __bf16* ob   = (__bf16*)ws; ws += (size_t)512 * 512 * 2;     // attn out

  ln_cast_kernel<<<65536, 256, 0, stream>>>(x, g1, b1, xn);
  ln_cast_kernel<<<512, 256, 0, stream>>>(lat, g2, b2, lnb);
  cast_kernel<<<(1024 * 512 / 4 + 255) / 256, 256, 0, stream>>>(Wq, wqb, 1024 * 512 / 4);
  cast_kernel<<<(1024 * 1024 / 4 + 255) / 256, 256, 0, stream>>>(Wkv, wkvb, 1024 * 1024 / 4);
  cast_kernel<<<(512 * 1024 / 4 + 255) / 256, 256, 0, stream>>>(Wout, wob, 512 * 1024 / 4);

  // kv = xn @ Wkv : [65536,1024] x [1024,1024]
  gemm_wmma_kernel<false><<<dim3(1024 / 64, 65536 / 64), 128, 0, stream>>>(
      xn, wkvb, kvb, 65536, 1024, 1024);
  // q = ln @ Wq : [512,1024] x [1024,512]
  gemm_wmma_kernel<false><<<dim3(512 / 64, 512 / 64), 128, 0, stream>>>(
      lnb, wqb, qb, 512, 512, 1024);
  // flash attention per (b,h)
  attn_kernel<<<64, 128, 0, stream>>>(qb, kvb, ob);
  // out = O @ Wout : [512,512] x [512,1024] -> fp32
  gemm_wmma_kernel<true><<<dim3(1024 / 64, 512 / 64), 128, 0, stream>>>(
      ob, wob, out, 512, 1024, 512);
}
